// RobustPPRGo_45870250721440
// MI455X (gfx1250) — matrix operand order repaired and checked by reading, not compiled, not run on hardware
//
#include <hip/hip_runtime.h>
#include <math.h>

// ---------------------------------------------------------------------------
// Types for CDNA5 WMMA (wave32): v_wmma_f32_16x16x32_bf16
// ---------------------------------------------------------------------------
typedef __attribute__((ext_vector_type(16))) __bf16 v16bf;
typedef __attribute__((ext_vector_type(8)))  __bf16 v8bf;
typedef __attribute__((ext_vector_type(8)))  float  v8f;

__device__ __forceinline__ unsigned short f2bf_bits(float x) {
  unsigned u = __builtin_bit_cast(unsigned, x);
  u += 0x7fffu + ((u >> 16) & 1u);            // round-to-nearest-even
  return (unsigned short)(u >> 16);
}
__device__ __forceinline__ __bf16 bits2bf(unsigned short h) {
  return __builtin_bit_cast(__bf16, h);
}

// Problem constants (from the reference)
#define NN   65536
#define FF   512
#define BB   2048
#define HH   256
#define CC   64
#define TOPK 32

// ---------------------------------------------------------------------------
// f32 -> bf16 conversion (weights)
// ---------------------------------------------------------------------------
__global__ void cvt_f32_bf16(const float* __restrict__ in,
                             unsigned short* __restrict__ out, int n) {
  int i = blockIdx.x * blockDim.x + threadIdx.x;
  if (i < n) out[i] = f2bf_bits(in[i]);
}

// ---------------------------------------------------------------------------
// Tiled bf16 WMMA GEMM:  C[M,N] = act(A[M,K] @ B[K,N])
//   block tile 128x64, K step 32, 8 waves (4x2), each wave 32x32 (2x2 WMMA)
//   A: f32 (layer 1) or bf16 (layers 2/3); B: bf16; C: bf16 or f32
//   LDS double-buffered; B staged transposed so both fragment loads are
//   contiguous 16B ds_load_b128 chunks matching the ISA 16-bit VGPR striping.
// ---------------------------------------------------------------------------
template <bool A_IS_BF16, bool RELU, bool OUT_BF16>
__global__ __launch_bounds__(256)
void gemm_bf16_wmma(const void* __restrict__ Ap,
                    const unsigned short* __restrict__ Bp,
                    void* __restrict__ Cp, int M, int Nn, int Kk) {
  constexpr int BM = 128, BN = 64, BK = 32, LDT = BK + 8;  // pad: 40 elems
  __shared__ __align__(16) __bf16 sA[2][BM * LDT];
  __shared__ __align__(16) __bf16 sB[2][BN * LDT];

  const int tid  = threadIdx.x;
  const int lane = tid & 31;
  const int wid  = tid >> 5;
  const int bm0  = blockIdx.x * BM;
  const int bn0  = blockIdx.y * BN;
  const int wm   = (wid >> 1) * 32;
  const int wn   = (wid & 1) * 32;

  const float*  Af = (const float*)Ap;
  const __bf16* Ab = (const __bf16*)Ap;
  (void)M;

  v8f acc[2][2] = {};

  auto stage = [&](int k0, int buf) {
    // A tile 128x32: each thread copies 16 contiguous elements of a row
#pragma unroll
    for (int i = 0; i < 16; ++i) {
      int flat = tid * 16 + i;
      int r = flat >> 5, c = flat & 31;
      long gi = (long)(bm0 + r) * Kk + (k0 + c);
      sA[buf][r * LDT + c] = A_IS_BF16 ? Ab[gi] : bits2bf(f2bf_bits(Af[gi]));
    }
    // B tile 32x64, stored transposed [n][k] so fragments read contiguously
#pragma unroll
    for (int i = 0; i < 8; ++i) {
      int flat = tid * 8 + i;
      int r = flat >> 6, n = flat & 63;
      sB[buf][n * LDT + r] = bits2bf(Bp[(long)(k0 + r) * Nn + (bn0 + n)]);
    }
  };

  const int nK = Kk / BK;
  stage(0, 0);
  __syncthreads();

  for (int kt = 0; kt < nK; ++kt) {
    const int buf = kt & 1;
    if (kt + 1 < nK) stage((kt + 1) * BK, buf ^ 1);

    // ISA 16-bit fragment striping: lanes 0-15 hold K 0..7 & 16..23,
    // lanes 16-31 hold K 8..15 & 24..31 (A); analogous halves for B.
    const int rsel = lane & 15;
    const int ah   = (lane < 16) ? 0 : 8;    // A K-base
    const int bh   = (lane < 16) ? 0 : 8;    // B K-base (same striping as A)
    v16bf a[2], b[2];
#pragma unroll
    for (int t = 0; t < 2; ++t) {
      const __bf16* pa = &sA[buf][(wm + t * 16 + rsel) * LDT + ah];
      v8bf alo = *(const v8bf*)pa;
      v8bf ahi = *(const v8bf*)(pa + 16);
#pragma unroll
      for (int j = 0; j < 8; ++j) { a[t][j] = alo[j]; a[t][8 + j] = ahi[j]; }
      const __bf16* pb = &sB[buf][(wn + t * 16 + rsel) * LDT + bh];
      v8bf blo = *(const v8bf*)pb;
      v8bf bhi = *(const v8bf*)(pb + 16);
#pragma unroll
      for (int j = 0; j < 8; ++j) { b[t][j] = blo[j]; b[t][8 + j] = bhi[j]; }
    }
#pragma unroll
    for (int tm = 0; tm < 2; ++tm)
#pragma unroll
      for (int tn = 0; tn < 2; ++tn)
        acc[tm][tn] = __builtin_amdgcn_wmma_f32_16x16x32_bf16(
            false, a[tm], false, b[tn], (short)0, acc[tm][tn], false, false);

    __syncthreads();
  }

  // Epilogue: C/D layout — VGPR r: lanes 0-15 (M=r, N=lane),
  // lanes 16-31 (M=r+8, N=lane-16)
  const int moff = (lane < 16) ? 0 : 8;
  const int ncol = lane & 15;
#pragma unroll
  for (int tm = 0; tm < 2; ++tm)
#pragma unroll
    for (int tn = 0; tn < 2; ++tn)
#pragma unroll
      for (int r = 0; r < 8; ++r) {
        int row = bm0 + wm + tm * 16 + r + moff;
        int col = bn0 + wn + tn * 16 + ncol;
        float v = acc[tm][tn][r];
        if (RELU) v = fmaxf(v, 0.0f);
        long gi = (long)row * Nn + col;
        if (OUT_BF16) ((unsigned short*)Cp)[gi] = f2bf_bits(v);
        else          ((float*)Cp)[gi] = v;
      }
}

// ---------------------------------------------------------------------------
// Fused top-32 + row-sum: one 64-thread block per ppr row (2048 blocks).
// Each thread keeps a fully-unrolled sorted top-32 in registers over its
// 1024-element strided slice; LDS merge selects the global top-32.
// ---------------------------------------------------------------------------
__global__ __launch_bounds__(64)
void topk_rowsum(const float* __restrict__ A, float* __restrict__ tv,
                 int* __restrict__ ti, float* __restrict__ rowsum, int Ncols) {
  constexpr int T = 64;
  __shared__ float sval[T * TOPK];
  __shared__ int   sidx[T * TOPK];
  __shared__ float rsum[T];
  __shared__ float rmax[T];
  __shared__ int   rpos[T];

  const int b = blockIdx.x;
  const int t = threadIdx.x;
  const float* row = A + (long)b * Ncols;

  float lv[TOPK];
  int   li[TOPK];
#pragma unroll
  for (int i = 0; i < TOPK; ++i) { lv[i] = -3.4e38f; li[i] = -1; }
  float s = 0.0f;

  for (int c = t; c < Ncols; c += T) {
    if (c + 16 * T < Ncols) __builtin_prefetch(&row[c + 16 * T], 0, 0);
    float x = row[c];
    s += x;
    if (x > lv[TOPK - 1]) {                 // beats current smallest kept
#pragma unroll
      for (int i = TOPK - 1; i >= 1; --i) { // sorted-descending insert
        if (x > lv[i - 1])    { lv[i] = lv[i - 1]; li[i] = li[i - 1]; }
        else if (x > lv[i])   { lv[i] = x;         li[i] = c; }
      }
      if (x > lv[0]) { lv[0] = x; li[0] = c; }
    }
  }

  rsum[t] = s;
#pragma unroll
  for (int i = 0; i < TOPK; ++i) { sval[t * TOPK + i] = lv[i]; sidx[t * TOPK + i] = li[i]; }
  __syncthreads();

  if (t == 0) {
    float tot = 0.0f;
    for (int i = 0; i < T; ++i) tot += rsum[i];
    rowsum[b] = tot;
  }

  for (int j = 0; j < TOPK; ++j) {
    float m = -3.4e38f; int mp = t * TOPK;
    for (int i = 0; i < TOPK; ++i) {
      float v = sval[t * TOPK + i];
      if (v > m) { m = v; mp = t * TOPK + i; }
    }
    rmax[t] = m; rpos[t] = mp;
    __syncthreads();
    if (t == 0) {
      float bm = -3.4e38f; int bp = 0;
      for (int i = 0; i < T; ++i)
        if (rmax[i] > bm) { bm = rmax[i]; bp = rpos[i]; }
      tv[b * TOPK + j] = bm;
      ti[b * TOPK + j] = sidx[bp];
      sval[bp] = -3.4e38f;
    }
    __syncthreads();
  }
}

// ---------------------------------------------------------------------------
// Soft-k-medoid: one 256-thread block per row. Gather 32x64 embeddings into
// LDS, 1024 pairwise L2 distances (4/thread) accumulated with ds_add_f32,
// serial 32-way softmax + weight correction, 64-wide weighted combine.
// ---------------------------------------------------------------------------
__global__ __launch_bounds__(256)
void soft_k_medoid(const float* __restrict__ logits, const float* __restrict__ tv,
                   const int* __restrict__ ti, const float* __restrict__ rowsum,
                   float* __restrict__ out) {
  __shared__ float xk[TOPK][CC];   // 8 KB
  __shared__ float vals[TOPK];
  __shared__ float dist[TOPK];
  __shared__ float w[TOPK];
  __shared__ int   idx[TOPK];

  const int b = blockIdx.x;
  const int t = threadIdx.x;

  if (t < TOPK) {
    vals[t] = tv[b * TOPK + t];
    idx[t]  = ti[b * TOPK + t];
    dist[t] = 0.0f;
  }
  __syncthreads();

#pragma unroll
  for (int i = 0; i < 8; ++i) {        // 2048 gathered floats
    int flat = t * 8 + i;
    int k = flat >> 6, ch = flat & 63;
    xk[k][ch] = logits[(long)idx[k] * CC + ch];
  }
  __syncthreads();

#pragma unroll
  for (int pi = 0; pi < 4; ++pi) {     // 1024 (c,m) pairs
    int p = t + pi * 256;
    int c = p >> 5, m = p & 31;
    float d2 = 0.0f;
#pragma unroll
    for (int ch = 0; ch < CC; ++ch) {
      float d = xk[c][ch] - xk[m][ch];
      d2 += d * d;
    }
    float l2 = sqrtf(d2 + 1e-12f);
    atomicAdd(&dist[c], vals[m] * l2);  // ds_add_f32
  }
  __syncthreads();

  if (t == 0) {                         // softmax(-dist) * vals, renormalized
    float dd[TOPK], mx = -3.4e38f;
    for (int i = 0; i < TOPK; ++i) {
      float di = (vals[i] == 0.0f) ? 3.4028235e38f : dist[i];
      dd[i] = -di;                      // temperature = 1
      if (dd[i] > mx) mx = dd[i];
    }
    float ssum = 0.0f;
    for (int i = 0; i < TOPK; ++i) {
      float e = __expf(dd[i] - mx) * vals[i];
      dd[i] = e; ssum += e;
    }
    float inv = 1.0f / ssum;
    for (int i = 0; i < TOPK; ++i) w[i] = dd[i] * inv;
  }
  __syncthreads();

  if (t < CC) {
    float acc = 0.0f;
#pragma unroll
    for (int k = 0; k < TOPK; ++k) acc += w[k] * xk[k][t];
    out[(long)b * CC + t] = rowsum[b] * acc;
  }
}

// ---------------------------------------------------------------------------
// Launch
// ---------------------------------------------------------------------------
extern "C" void kernel_launch(void* const* d_in, const int* in_sizes, int n_in,
                              void* d_out, int out_size, void* d_ws, size_t ws_size,
                              hipStream_t stream) {
  (void)in_sizes; (void)n_in; (void)out_size; (void)ws_size;

  const float* X   = (const float*)d_in[0];   // [N,F]
  const float* PPR = (const float*)d_in[1];   // [B,N]
  const float* W1  = (const float*)d_in[2];   // [F,H]
  const float* W2  = (const float*)d_in[3];   // [H,H]
  const float* W3  = (const float*)d_in[4];   // [H,C]

  char* ws = (char*)d_ws;
  size_t o = 0;
  unsigned short* w1b = (unsigned short*)(ws + o); o += (size_t)FF * HH * 2;
  unsigned short* w2b = (unsigned short*)(ws + o); o += (size_t)HH * HH * 2;
  unsigned short* w3b = (unsigned short*)(ws + o); o += (size_t)HH * CC * 2;
  unsigned short* h1  = (unsigned short*)(ws + o); o += (size_t)NN * HH * 2;
  unsigned short* h2  = (unsigned short*)(ws + o); o += (size_t)NN * HH * 2;
  float*          lg  = (float*)(ws + o);          o += (size_t)NN * CC * 4;
  float*          tv  = (float*)(ws + o);          o += (size_t)BB * TOPK * 4;
  int*            ti  = (int*)(ws + o);            o += (size_t)BB * TOPK * 4;
  float*          rs  = (float*)(ws + o);          o += (size_t)BB * 4;

  // 1) weights -> bf16
  cvt_f32_bf16<<<(FF * HH + 255) / 256, 256, 0, stream>>>(W1, w1b, FF * HH);
  cvt_f32_bf16<<<(HH * HH + 255) / 256, 256, 0, stream>>>(W2, w2b, HH * HH);
  cvt_f32_bf16<<<(HH * CC + 255) / 256, 256, 0, stream>>>(W3, w3b, HH * CC);

  // 2) MLP: h1 = relu(X@W1), h2 = relu(h1@W2), logits = h2@W3
  gemm_bf16_wmma<false, true,  true ><<<dim3(NN / 128, HH / 64), 256, 0, stream>>>(
      (const void*)X,  w1b, (void*)h1, NN, HH, FF);
  gemm_bf16_wmma<true,  true,  true ><<<dim3(NN / 128, HH / 64), 256, 0, stream>>>(
      (const void*)h1, w2b, (void*)h2, NN, HH, HH);
  gemm_bf16_wmma<true,  false, false><<<dim3(NN / 128, CC / 64), 256, 0, stream>>>(
      (const void*)h2, w3b, (void*)lg, NN, CC, HH);

  // 3) top-32 + row sums over ppr_scores
  topk_rowsum<<<BB, 64, 0, stream>>>(PPR, tv, ti, rs, NN);

  // 4) soft k-medoid -> d_out [B, C] f32
  soft_k_medoid<<<BB, 256, 0, stream>>>(lg, tv, ti, rs, (float*)d_out);
}
